// LocalEnvironmentEmbedding_11605001634318
// MI455X (gfx1250) — matrix operand order repaired and checked by reading, not compiled, not run on hardware
//
#include <hip/hip_runtime.h>

typedef __attribute__((ext_vector_type(16))) _Float16 v16h;
typedef __attribute__((ext_vector_type(8)))  float    v8f;

#define WAVES 8
#define TPB (WAVES * 32)
#define NFRAG 48                 // 24 (W_lin) + 8*3 (W1..W3)
#define FRAG_HALFS (32 * 16)     // 32 lanes x 16 f16 = 512 halves = 1KB/frag

__device__ __forceinline__ float silu_n(float x) {
    // silu(x) * 1.679177
    return (x / (1.0f + __expf(-x))) * 1.679177f;
}

// Load 16 consecutive f32 and convert to a 16xf16 B-operand fragment half-row.
__device__ __forceinline__ v16h cvt_row16(const float* __restrict__ p) {
    float4 f0 = *(const float4*)(p);
    float4 f1 = *(const float4*)(p + 4);
    float4 f2 = *(const float4*)(p + 8);
    float4 f3 = *(const float4*)(p + 12);
    v16h b;
    b[0]  = (_Float16)f0.x; b[1]  = (_Float16)f0.y; b[2]  = (_Float16)f0.z; b[3]  = (_Float16)f0.w;
    b[4]  = (_Float16)f1.x; b[5]  = (_Float16)f1.y; b[6]  = (_Float16)f1.z; b[7]  = (_Float16)f1.w;
    b[8]  = (_Float16)f2.x; b[9]  = (_Float16)f2.y; b[10] = (_Float16)f2.z; b[11] = (_Float16)f2.w;
    b[12] = (_Float16)f3.x; b[13] = (_Float16)f3.y; b[14] = (_Float16)f3.z; b[15] = (_Float16)f3.w;
    return b;
}

// Convert two f32 D-accumulator tiles (G[2kt], G[2kt+1]) into the next GEMM's
// f16 B fragment. D tile: VGPR v holds (m = v + 8*h, n = lane&15).
// B frag needs lane elem i <-> k = 16*h + i. Only cross-lane movement needed is
// the half swap between lane pairs (l, l^16).
template<bool ACT>
__device__ __forceinline__ v16h dtob(const v8f& Ga, const v8f& Gb, int h) {
    v16h b;
#pragma unroll
    for (int v = 0; v < 8; ++v) {
        float own  = h ? Gb[v] : Ga[v];           // tile 2kt+h, m_local = v + 8h
        float send = h ? Ga[v] : Gb[v];           // what the partner lane needs
        float recv = __shfl_xor(send, 16, 32);    // wave32 half swap
        float lo = h ? recv : own;                // element v      (k = 16h + v)
        float hi = h ? own  : recv;               // element 8 + v  (k = 16h + 8 + v)
        if (ACT) { lo = silu_n(lo); hi = silu_n(hi); }
        b[v]     = (_Float16)lo;
        b[8 + v] = (_Float16)hi;
    }
    return b;
}

__device__ __forceinline__ void gemm64(const _Float16* s_w, int fragBase,
                                       const v16h b[2], v8f acc[4], int lane) {
#pragma unroll
    for (int mt = 0; mt < 4; ++mt) {
        v8f c = {};
#pragma unroll
        for (int kt = 0; kt < 2; ++kt) {
            v16h a = *(const v16h*)(s_w + ((fragBase + mt * 2 + kt) * 32 + lane) * 16);
            c = __builtin_amdgcn_wmma_f32_16x16x32_f16(false, a, false, b[kt],
                                                       (short)0, c, false, false);
        }
        acc[mt] = c;
    }
}

// Outer-product emit for one 16-wide weight group with block dim D.
// Lane covers j = 8h..8h+7, i.e. a contiguous run of 8*D output floats,
// all runs start at multiples of 4 floats -> pure float4 stores.
template<int D, int AOFF, int OUTOFF>
__device__ __forceinline__ void emit_group(const v8f& w, const float ea[16],
                                           float* __restrict__ op, int h) {
    float* o = op + OUTOFF + h * 8 * D;
#pragma unroll
    for (int q = 0; q < 2 * D; ++q) {
        float4 f;
        f.x = w[(4 * q + 0) / D] * ea[AOFF + (4 * q + 0) % D];
        f.y = w[(4 * q + 1) / D] * ea[AOFF + (4 * q + 1) % D];
        f.z = w[(4 * q + 2) / D] * ea[AOFF + (4 * q + 2) % D];
        f.w = w[(4 * q + 3) / D] * ea[AOFF + (4 * q + 3) % D];
        *(float4*)(o + 4 * q) = f;
    }
}

__global__ __launch_bounds__(TPB) void lee_gfx1250_kernel(
    const int*   __restrict__ edge_index,   // [2, E] (src row then dst row)
    const float* __restrict__ node_attr,    // [N, 64]
    const float* __restrict__ edge_attr,    // [E, 16]
    const float* __restrict__ edge_embed,   // [E, 64]
    const float* __restrict__ W_lin,        // [192, 64]
    const float* __restrict__ W1,           // [64, 64]
    const float* __restrict__ W2,
    const float* __restrict__ W3,
    float* __restrict__ out,                // [E, 256]
    int E)
{
    __shared__ __align__(32) _Float16 s_w[NFRAG * FRAG_HALFS];  // 48 KB

    // ---- cooperative fill: weights, pre-scaled, transposed (A[m][k] = W[k][m]),
    //      swizzled into exact A-fragment layout (f16) ----
    for (int idx = threadIdx.x; idx < NFRAG * FRAG_HALFS; idx += TPB) {
        int frag = idx >> 9;      // /512 halves per frag
        int rem  = idx & 511;
        int lane = rem >> 4;
        int j    = rem & 15;
        int khalf = lane >> 4;
        int mloc  = lane & 15;
        int mt, kt; const float* W; float scale;
        if (frag < 24) {                       // W_lin: mt in 0..3, kt in 0..5
            W = W_lin; scale = 0.0721687836f;  // 1/sqrt(192)
            mt = frag / 6; kt = frag % 6;
        } else {
            int loc = frag - 24; int which = loc >> 3; int r8 = loc & 7;
            mt = r8 >> 1; kt = r8 & 1; scale = 0.125f;
            W = (which == 0) ? W1 : (which == 1) ? W2 : W3;
        }
        // A 16x32 f16 frag, contiguous k order used here (not ISA table order): elem j -> k = 16*khalf + j
        int k = kt * 32 + khalf * 16 + j;
        int m = mt * 16 + mloc;
        s_w[idx] = (_Float16)(W[k * 64 + m] * scale);
    }
    __syncthreads();

    const int lane = threadIdx.x & 31;
    const int wv   = threadIdx.x >> 5;
    const int h    = lane >> 4;     // K-half of the B operand this lane carries
    const int r    = lane & 15;     // edge row within the 16-edge tile
    const int numTiles = (E + 15) >> 4;

    for (int tile = blockIdx.x * WAVES + wv; tile < numTiles;
         tile += gridDim.x * WAVES) {
        int e = tile * 16 + r;
        e = min(e, E - 1);          // tail lanes duplicate the last edge (benign)
        int si = edge_index[e];
        int di = edge_index[E + e];
        const float* sp = node_attr  + (size_t)si * 64;
        const float* dp = node_attr  + (size_t)di * 64;
        const float* ep = edge_embed + (size_t)e  * 64;

        // ---- GEMM1: G1 = W_lin^T (64x192) x feats^T (192x16) ----
        v16h bf[6];
#pragma unroll
        for (int kt = 0; kt < 6; ++kt) {
            const float* base = (kt < 2) ? sp : (kt < 4) ? dp : ep;
            int co = ((kt & 1) << 5) + (h << 4);   // 16 consecutive floats
            bf[kt] = cvt_row16(base + co);
        }
        v8f acc[4];
#pragma unroll
        for (int mt = 0; mt < 4; ++mt) {
            v8f c = {};
#pragma unroll
            for (int kt = 0; kt < 6; ++kt) {
                v16h a = *(const v16h*)(s_w + ((mt * 6 + kt) * 32 + lane) * 16);
                c = __builtin_amdgcn_wmma_f32_16x16x32_f16(false, a, false, bf[kt],
                                                           (short)0, c, false, false);
            }
            acc[mt] = c;
        }

        // ---- GEMM2 (W1): no activation on input, silu applied after ----
        v16h bh[2];
        bh[0] = dtob<false>(acc[0], acc[1], h);
        bh[1] = dtob<false>(acc[2], acc[3], h);
        gemm64(s_w, 24, bh, acc, lane);

        // ---- GEMM3 (W2) ----
        bh[0] = dtob<true>(acc[0], acc[1], h);
        bh[1] = dtob<true>(acc[2], acc[3], h);
        gemm64(s_w, 32, bh, acc, lane);

        // ---- GEMM4 (W3) -> weights ----
        bh[0] = dtob<true>(acc[0], acc[1], h);
        bh[1] = dtob<true>(acc[2], acc[3], h);
        gemm64(s_w, 40, bh, acc, lane);

        // ---- outer product with edge_attr, blocks d = 1,3,5,7 ----
        const float* eap = edge_attr + (size_t)e * 16;
        float4 a0 = *(const float4*)(eap);
        float4 a1 = *(const float4*)(eap + 4);
        float4 a2 = *(const float4*)(eap + 8);
        float4 a3 = *(const float4*)(eap + 12);
        float ea[16] = {a0.x, a0.y, a0.z, a0.w, a1.x, a1.y, a1.z, a1.w,
                        a2.x, a2.y, a2.z, a2.w, a3.x, a3.y, a3.z, a3.w};
        float* op = out + (size_t)e * 256;
        emit_group<1, 0,   0>(acc[0], ea, op, h);
        emit_group<3, 1,  16>(acc[1], ea, op, h);
        emit_group<5, 4,  64>(acc[2], ea, op, h);
        emit_group<7, 9, 144>(acc[3], ea, op, h);
    }
}

extern "C" void kernel_launch(void* const* d_in, const int* in_sizes, int n_in,
                              void* d_out, int out_size, void* d_ws, size_t ws_size,
                              hipStream_t stream) {
    const int*   edge_index = (const int*)d_in[0];
    const float* node_attr  = (const float*)d_in[1];
    const float* edge_attr  = (const float*)d_in[2];
    const float* edge_embed = (const float*)d_in[3];
    const float* W_lin      = (const float*)d_in[4];
    const float* W1         = (const float*)d_in[5];
    const float* W2         = (const float*)d_in[6];
    const float* W3         = (const float*)d_in[7];
    float* out = (float*)d_out;

    int E = in_sizes[2] / 16;               // edge_attr is [E,16]
    int numTiles = (E + 15) / 16;
    int blocks = (numTiles + WAVES - 1) / WAVES;
    if (blocks > 1024) blocks = 1024;
    if (blocks < 1) blocks = 1;

    lee_gfx1250_kernel<<<blocks, TPB, 0, stream>>>(
        edge_index, node_attr, edge_attr, edge_embed,
        W_lin, W1, W2, W3, out, E);
}
